// MixtureOfAdapterWithClassifier_42236708389581
// MI455X (gfx1250) — matrix operand order repaired and checked by reading, not compiled, not run on hardware
//
#include <hip/hip_runtime.h>

typedef __bf16 bf16;
typedef __attribute__((ext_vector_type(16))) __bf16 v16bf;
typedef __attribute__((ext_vector_type(8)))  __bf16 v8bf;
typedef __attribute__((ext_vector_type(8)))  float  v8f;

#define EPSV 1e-6f
#define NEGV -1000000000.0f

// Problem dims (fixed by reference)
#define NTOK 8192   // B*L
#define HDIM 1024
#define FDIM 2048

// GEMM tiling
#define BM 128
#define BN 128
#define BK 32
#define LDSS 40     // padded LDS row stride in bf16 elements (80B, 16B aligned)

// ---------------------------------------------------------------------------
// CDNA5 async global->LDS copy (ASYNCcnt) helpers
// ---------------------------------------------------------------------------
__device__ __forceinline__ unsigned lds_off(const void* p) {
    // generic pointer to LDS: low 32 bits are the wave-relative LDS offset
    return (unsigned)(size_t)p;
}
__device__ __forceinline__ void async_copy_b128(unsigned loff, unsigned long long g) {
    asm volatile("global_load_async_to_lds_b128 %0, %1, off"
                 :: "v"(loff), "v"(g) : "memory");
}
__device__ __forceinline__ void wait_async0() {
    asm volatile("s_wait_asynccnt 0x0" ::: "memory");
}

// ---------------------------------------------------------------------------
// Kernel 1: per-token gate MLP + softmax + LayerNorm statistics (one wave/token)
// ---------------------------------------------------------------------------
__global__ __launch_bounds__(256) void gate_stats_kernel(
    const float* __restrict__ x, const int* __restrict__ dmask,
    const float* __restrict__ gw1, const float* __restrict__ gb1,
    const float* __restrict__ gw2, const float* __restrict__ gb2,
    float* __restrict__ gateW, float* __restrict__ meanv, float* __restrict__ rstdv)
{
    int wave = threadIdx.x >> 5;
    int lane = threadIdx.x & 31;
    int t = blockIdx.x * 8 + wave;
    if (t >= NTOK) return;
    const float* xr = x + (size_t)t * HDIM;

    float s = 0.f, ss = 0.f, h0 = 0.f, h1 = 0.f, h2 = 0.f, h3 = 0.f;
    for (int i = lane; i < HDIM; i += 32) {
        float xv = xr[i];
        s += xv; ss += xv * xv;
        float4 g = ((const float4*)gw1)[i];
        h0 += xv * g.x; h1 += xv * g.y; h2 += xv * g.z; h3 += xv * g.w;
    }
    #pragma unroll
    for (int m = 16; m >= 1; m >>= 1) {
        s  += __shfl_xor(s,  m, 32);  ss += __shfl_xor(ss, m, 32);
        h0 += __shfl_xor(h0, m, 32);  h1 += __shfl_xor(h1, m, 32);
        h2 += __shfl_xor(h2, m, 32);  h3 += __shfl_xor(h3, m, 32);
    }
    if (lane == 0) {
        float mean = s * (1.f / HDIM);
        float var  = ss * (1.f / HDIM) - mean * mean;
        float rstd = rsqrtf(var + EPSV);
        float hv[4];
        hv[0] = fmaxf(h0 + gb1[0], 0.f);
        hv[1] = fmaxf(h1 + gb1[1], 0.f);
        hv[2] = fmaxf(h2 + gb1[2], 0.f);
        hv[3] = fmaxf(h3 + gb1[3], 0.f);
        float lg[4];
        #pragma unroll
        for (int e = 0; e < 4; e++) {
            float acc = gb2[e];
            #pragma unroll
            for (int d = 0; d < 4; d++) acc += hv[d] * gw2[d * 4 + e];
            lg[e] = (dmask[e] == 0) ? NEGV : acc;
        }
        float mx = fmaxf(fmaxf(lg[0], lg[1]), fmaxf(lg[2], lg[3]));
        float p[4], sum = 0.f;
        #pragma unroll
        for (int e = 0; e < 4; e++) { p[e] = __expf(lg[e] - mx); sum += p[e]; }
        float inv = 1.f / sum;
        gateW[t * 3 + 0] = p[0] * inv;
        gateW[t * 3 + 1] = p[1] * inv;
        gateW[t * 3 + 2] = p[2] * inv;
        meanv[t] = mean;
        rstdv[t] = rstd;
    }
}

// ---------------------------------------------------------------------------
// Kernel 2: transpose + f32->bf16: src f32 [R][C] -> dst bf16 [C][R]
// ---------------------------------------------------------------------------
__global__ __launch_bounds__(256) void transpose_cvt_kernel(
    const float* __restrict__ src, bf16* __restrict__ dst, int R, int C)
{
    __shared__ float tile[32][33];
    int bx = blockIdx.x * 32;   // C base
    int by = blockIdx.y * 32;   // R base
    int tx = threadIdx.x;       // 0..31
    int ty = threadIdx.y;       // 0..7
    #pragma unroll
    for (int j = 0; j < 32; j += 8)
        tile[ty + j][tx] = src[(size_t)(by + ty + j) * C + bx + tx];
    __syncthreads();
    #pragma unroll
    for (int j = 0; j < 32; j += 8)
        dst[(size_t)(bx + ty + j) * R + by + tx] = (bf16)tile[tx][ty + j];
}

// ---------------------------------------------------------------------------
// Kernel 3: out = x * (1 + w0)
// ---------------------------------------------------------------------------
__global__ __launch_bounds__(256) void init_out_kernel(
    const float* __restrict__ x, const float* __restrict__ gateW,
    float* __restrict__ out, int n4)
{
    int i = blockIdx.x * blockDim.x + threadIdx.x;
    if (i < n4) {
        int tok = (int)(((size_t)i * 4) / HDIM);
        float w0 = 1.f + gateW[tok * 3];
        float4 v = ((const float4*)x)[i];
        v.x *= w0; v.y *= w0; v.z *= w0; v.w *= w0;
        ((float4*)out)[i] = v;
    }
}

// ---------------------------------------------------------------------------
// WMMA fragment loaders (per CDNA5 ISA 16-bit layouts)
// ---------------------------------------------------------------------------
__device__ __forceinline__ v16bf load_frag_A(const bf16* As, int row, int lane)
{
    // lane<16: K elems {0..7, 16..23}; lane>=16: K elems {8..15, 24..31}
    const bf16* p = As + row * LDSS + ((lane & 16) ? 8 : 0);
    v8bf lo = *(const v8bf*)p;
    v8bf hi = *(const v8bf*)(p + 16);
    v16bf r;
    #pragma unroll
    for (int e = 0; e < 8; e++) { r[e] = lo[e]; r[e + 8] = hi[e]; }
    return r;
}

__device__ __forceinline__ v16bf load_frag_B(const bf16* Bs, int colBase, int lane)
{
    // Bs transposed in LDS: Bs[n][k]. lane<16: K {0..7, 16..23}; lane>=16: K {8..15, 24..31}
    const bf16* p = Bs + (colBase + (lane & 15)) * LDSS + ((lane & 16) ? 8 : 0);
    v8bf lo = *(const v8bf*)p;
    v8bf hi = *(const v8bf*)(p + 16);
    v16bf r;
    #pragma unroll
    for (int e = 0; e < 8; e++) { r[e] = lo[e]; r[e + 8] = hi[e]; }
    return r;
}

// ---------------------------------------------------------------------------
// Kernel 4: GEMM1  Y = relu(LN(x) @ W1 + b1)  [M=8192, K=HDIM, N=FDIM]
// A: LN applied on the fly (manual LDS staging). B: async global->LDS from W1^T.
// Double-buffered LDS pipeline.
// ---------------------------------------------------------------------------
__global__ __launch_bounds__(256) void adapter_gemm1_kernel(
    const float* __restrict__ x, const float* __restrict__ meanv, const float* __restrict__ rstdv,
    const float* __restrict__ ln_s, const float* __restrict__ ln_b,
    const bf16* __restrict__ W1T,  // [FDIM][HDIM]
    const float* __restrict__ b1,
    bf16* __restrict__ Y)
{
    __shared__ __attribute__((aligned(16))) bf16 As[2][BM * LDSS];
    __shared__ __attribute__((aligned(16))) bf16 Bs[2][BN * LDSS];

    int tid = threadIdx.x;
    int lane = tid & 31, wave = tid >> 5;
    int wm = wave & 1, wn = wave >> 1;          // 2x4 wave grid
    int m0 = blockIdx.y * BM;
    int n0 = blockIdx.x * BN;

    v8f zero;
    #pragma unroll
    for (int e = 0; e < 8; e++) zero[e] = 0.f;
    v8f acc[4][2];
    #pragma unroll
    for (int mi = 0; mi < 4; mi++)
        #pragma unroll
        for (int ni = 0; ni < 2; ni++) acc[mi][ni] = zero;

    int arow  = tid >> 1;        // 0..127 (row of A tile and of B^T tile)
    int ahalf = tid & 1;         // 16-col half
    float amean = meanv[m0 + arow];
    float arstd = rstdv[m0 + arow];
    const float* xrow = x + (size_t)(m0 + arow) * HDIM + ahalf * 16;
    const bf16* brow  = W1T + (size_t)(n0 + arow) * HDIM + ahalf * 16;

    auto stage = [&](int k0, int buf) {
        // ---- B tile: plain copy -> async DMA to LDS (2 x 16B per thread) ----
        unsigned bl = lds_off(&Bs[buf][arow * LDSS + ahalf * 16]);
        unsigned long long bg = (unsigned long long)(brow + k0);
        async_copy_b128(bl, bg);
        async_copy_b128(bl + 16, bg + 16);
        // ---- A tile: LN + bf16 convert on the fly ----
        const float* xp = xrow + k0;
        const float* sp = ln_s + k0 + ahalf * 16;
        const float* bp = ln_b + k0 + ahalf * 16;
        bf16* dst = &As[buf][arow * LDSS + ahalf * 16];
        #pragma unroll
        for (int j = 0; j < 16; j += 4) {
            float4 v  = *(const float4*)(xp + j);
            float4 sv = *(const float4*)(sp + j);
            float4 bv = *(const float4*)(bp + j);
            dst[j + 0] = (bf16)(((v.x - amean) * arstd) * sv.x + bv.x);
            dst[j + 1] = (bf16)(((v.y - amean) * arstd) * sv.y + bv.y);
            dst[j + 2] = (bf16)(((v.z - amean) * arstd) * sv.z + bv.z);
            dst[j + 3] = (bf16)(((v.w - amean) * arstd) * sv.w + bv.w);
        }
        if (k0 + 2 * BK < HDIM) __builtin_prefetch(xp + 2 * BK, 0, 0);
    };

    stage(0, 0);
    wait_async0();
    __syncthreads();

    for (int k0 = 0; k0 < HDIM; k0 += BK) {
        int cur = (k0 >> 5) & 1;
        if (k0 + BK < HDIM) stage(k0 + BK, cur ^ 1);

        v16bf af[4], bfg[2];
        #pragma unroll
        for (int mi = 0; mi < 4; mi++)
            af[mi] = load_frag_A(&As[cur][0], wm * 64 + mi * 16 + (lane & 15), lane);
        #pragma unroll
        for (int ni = 0; ni < 2; ni++)
            bfg[ni] = load_frag_B(&Bs[cur][0], wn * 32 + ni * 16, lane);

        #pragma unroll
        for (int mi = 0; mi < 4; mi++)
            #pragma unroll
            for (int ni = 0; ni < 2; ni++)
                acc[mi][ni] = __builtin_amdgcn_wmma_f32_16x16x32_bf16(
                    false, af[mi], false, bfg[ni], (short)0, acc[mi][ni], false, false);

        wait_async0();
        __syncthreads();
    }

    // ---- epilogue: relu(acc + b1) -> bf16 Y ----
    int mrb  = (lane >> 4) * 8;   // 0 or 8
    int ncol = lane & 15;
    #pragma unroll
    for (int mi = 0; mi < 4; mi++) {
        #pragma unroll
        for (int ni = 0; ni < 2; ni++) {
            int gn = n0 + wn * 32 + ni * 16 + ncol;
            float bias = b1[gn];
            #pragma unroll
            for (int r = 0; r < 8; r++) {
                int gm = m0 + wm * 64 + mi * 16 + mrb + r;
                float v = fmaxf(acc[mi][ni][r] + bias, 0.f);
                Y[(size_t)gm * FDIM + gn] = (bf16)v;
            }
        }
    }
}

// ---------------------------------------------------------------------------
// Kernel 5: GEMM2  out += w[n][aidx] * (Y @ W2 + b2)  [M=8192, K=FDIM, N=HDIM]
// Both A and B tiles staged by async global->LDS DMA. Double buffered.
// ---------------------------------------------------------------------------
__global__ __launch_bounds__(256) void adapter_gemm2_kernel(
    const bf16* __restrict__ Y,
    const bf16* __restrict__ W2T,  // [HDIM][FDIM]
    const float* __restrict__ b2, const float* __restrict__ gateW,
    int aidx, float* __restrict__ out)
{
    __shared__ __attribute__((aligned(16))) bf16 As[2][BM * LDSS];
    __shared__ __attribute__((aligned(16))) bf16 Bs[2][BN * LDSS];

    int tid = threadIdx.x;
    int lane = tid & 31, wave = tid >> 5;
    int wm = wave & 1, wn = wave >> 1;
    int m0 = blockIdx.y * BM;
    int n0 = blockIdx.x * BN;

    v8f zero;
    #pragma unroll
    for (int e = 0; e < 8; e++) zero[e] = 0.f;
    v8f acc[4][2];
    #pragma unroll
    for (int mi = 0; mi < 4; mi++)
        #pragma unroll
        for (int ni = 0; ni < 2; ni++) acc[mi][ni] = zero;

    int arow  = tid >> 1;
    int ahalf = tid & 1;
    const bf16* yrow = Y   + (size_t)(m0 + arow) * FDIM + ahalf * 16;
    const bf16* brow = W2T + (size_t)(n0 + arow) * FDIM + ahalf * 16;

    auto stage = [&](int k0, int buf) {
        unsigned al = lds_off(&As[buf][arow * LDSS + ahalf * 16]);
        unsigned long long ag = (unsigned long long)(yrow + k0);
        async_copy_b128(al, ag);
        async_copy_b128(al + 16, ag + 16);
        unsigned bl = lds_off(&Bs[buf][arow * LDSS + ahalf * 16]);
        unsigned long long bg = (unsigned long long)(brow + k0);
        async_copy_b128(bl, bg);
        async_copy_b128(bl + 16, bg + 16);
    };

    stage(0, 0);
    wait_async0();
    __syncthreads();

    for (int k0 = 0; k0 < FDIM; k0 += BK) {
        int cur = (k0 >> 5) & 1;
        if (k0 + BK < FDIM) stage(k0 + BK, cur ^ 1);

        v16bf af[4], bfg[2];
        #pragma unroll
        for (int mi = 0; mi < 4; mi++)
            af[mi] = load_frag_A(&As[cur][0], wm * 64 + mi * 16 + (lane & 15), lane);
        #pragma unroll
        for (int ni = 0; ni < 2; ni++)
            bfg[ni] = load_frag_B(&Bs[cur][0], wn * 32 + ni * 16, lane);

        #pragma unroll
        for (int mi = 0; mi < 4; mi++)
            #pragma unroll
            for (int ni = 0; ni < 2; ni++)
                acc[mi][ni] = __builtin_amdgcn_wmma_f32_16x16x32_bf16(
                    false, af[mi], false, bfg[ni], (short)0, acc[mi][ni], false, false);

        wait_async0();
        __syncthreads();
    }

    // ---- epilogue: out += w[token][aidx] * (acc + b2) ----
    int mrb  = (lane >> 4) * 8;
    int ncol = lane & 15;
    #pragma unroll
    for (int mi = 0; mi < 4; mi++) {
        #pragma unroll
        for (int ni = 0; ni < 2; ni++) {
            int gn = n0 + wn * 32 + ni * 16 + ncol;
            float bias = b2[gn];
            #pragma unroll
            for (int r = 0; r < 8; r++) {
                int gm = m0 + wm * 64 + mi * 16 + mrb + r;
                float wsel = gateW[gm * 3 + aidx];
                size_t oi = (size_t)gm * HDIM + gn;
                out[oi] += wsel * (acc[mi][ni][r] + bias);
            }
        }
    }
}

// ---------------------------------------------------------------------------
extern "C" void kernel_launch(void* const* d_in, const int* in_sizes, int n_in,
                              void* d_out, int out_size, void* d_ws, size_t ws_size,
                              hipStream_t stream) {
    const float* x     = (const float*)d_in[0];
    const int*   dmask = (const int*)d_in[1];
    const float* gw1   = (const float*)d_in[2];
    const float* gb1   = (const float*)d_in[3];
    const float* gw2   = (const float*)d_in[4];
    const float* gb2   = (const float*)d_in[5];
    const float* lnsb  = (const float*)d_in[6];
    const float* lnbb  = (const float*)d_in[7];
    const float* lnsi  = (const float*)d_in[8];
    const float* lnbi  = (const float*)d_in[9];
    const float* aw1   = (const float*)d_in[10];
    const float* ab1   = (const float*)d_in[11];
    const float* aw2   = (const float*)d_in[12];
    const float* ab2   = (const float*)d_in[13];
    float* out = (float*)d_out;

    char* ws = (char*)d_ws;
    size_t off = 0;
    auto alloc = [&](size_t bytes) {
        void* p = ws + off;
        off = (off + bytes + 255) & ~(size_t)255;
        return p;
    };
    float* gateW = (float*)alloc((size_t)NTOK * 3 * 4);
    float* meanv = (float*)alloc((size_t)NTOK * 4);
    float* rstdv = (float*)alloc((size_t)NTOK * 4);
    bf16*  W1T   = (bf16*)alloc((size_t)FDIM * HDIM * 2);  // [FDIM][HDIM]
    bf16*  W2T   = (bf16*)alloc((size_t)HDIM * FDIM * 2);  // [HDIM][FDIM]
    bf16*  Ybf   = (bf16*)alloc((size_t)NTOK * FDIM * 2);

    // 1) gate weights + LN stats (one pass over x)
    gate_stats_kernel<<<NTOK / 8, 256, 0, stream>>>(
        x, dmask, gw1, gb1, gw2, gb2, gateW, meanv, rstdv);

    // 2) weight transpose + bf16 conversion (one-time, amortized over K-reuse)
    {
        dim3 tb(32, 8);
        dim3 g1(FDIM / 32, HDIM / 32);   // aw1: [HDIM][FDIM] -> W1T [FDIM][HDIM]
        transpose_cvt_kernel<<<g1, tb, 0, stream>>>(aw1, W1T, HDIM, FDIM);
        dim3 g2(HDIM / 32, FDIM / 32);   // aw2: [FDIM][HDIM] -> W2T [HDIM][FDIM]
        transpose_cvt_kernel<<<g2, tb, 0, stream>>>(aw2, W2T, FDIM, HDIM);
    }

    // 3) out = x * (1 + w0)
    int no4 = NTOK * HDIM / 4;
    init_out_kernel<<<(no4 + 255) / 256, 256, 0, stream>>>(x, gateW, out, no4);

    dim3 blk(256);
    dim3 g1(FDIM / BN, NTOK / BM);   // (16, 64)
    dim3 g2(HDIM / BN, NTOK / BM);   // (8, 64)

    // adapter "book" (weight index 1)
    adapter_gemm1_kernel<<<g1, blk, 0, stream>>>(x, meanv, rstdv, lnsb, lnbb, W1T, ab1, Ybf);
    adapter_gemm2_kernel<<<g2, blk, 0, stream>>>(Ybf, W2T, ab2, gateW, 1, out);

    // adapter "iwslt" (weight index 2)
    adapter_gemm1_kernel<<<g1, blk, 0, stream>>>(x, meanv, rstdv, lnsi, lnbi, W1T, ab1, Ybf);
    adapter_gemm2_kernel<<<g2, blk, 0, stream>>>(Ybf, W2T, ab2, gateW, 2, out);
}
